// MultiHeadAttnRelPos_46866683134052
// MI455X (gfx1250) — matrix operand order repaired and checked by reading, not compiled, not run on hardware
//
#include <hip/hip_runtime.h>
#include <hip/hip_bf16.h>
#include <stdint.h>

typedef __attribute__((ext_vector_type(16))) _Float16 v16h;
typedef __attribute__((ext_vector_type(8)))  float    v8f;

#define TOT_LEN 1024
#define MEM_LEN 1024
#define SEG_LEN 2048
#define BSZ     4
#define D_EMBED 1024
#define N_HEAD  16
#define D_HEAD  64

// A-fragment K index (16x32 f16 A): K = e + 16*half (matches B)
__device__ __forceinline__ int a_kk(int e, int half) {
    return e + 16 * half;
}
// B-fragment K index (32x16 f16 B): K = e + 16*half, lane holds column n = lane&15
__device__ __forceinline__ int b_kk(int e, int half) { return e + 16 * half; }

// ---------------------------------------------------------------------------
// Tiled f16-WMMA GEMM:  C(MxN) = A(MxK, fp32, row-split across A0/A1) * B(KxN, fp32)
// Block: 128 threads (4 waves). 64x64 C tile per block; wave w owns rows w*16..w*16+15.
// ---------------------------------------------------------------------------
template<bool OUT_HALF>
__global__ __launch_bounds__(128) void gemm_f16_kernel(
    const float* __restrict__ A0, const float* __restrict__ A1, int Msplit,
    const float* __restrict__ B, void* __restrict__ Cout,
    int M, int N, int K)
{
    __shared__ _Float16 As[64][40];   // 80B row stride (16B aligned)
    __shared__ _Float16 Bs[32][72];

    const int tid  = threadIdx.x;
    const int lane = tid & 31;
    const int wave = tid >> 5;
    const int mloc = lane & 15;
    const int half = lane >> 4;
    const int rb   = blockIdx.y * 64;
    const int cb   = blockIdx.x * 64;

    v8f acc[4];
    #pragma unroll
    for (int ct = 0; ct < 4; ++ct)
        #pragma unroll
        for (int v = 0; v < 8; ++v) acc[ct][v] = 0.0f;

    const int arow  = tid >> 1;        // 0..63
    const int acol0 = (tid & 1) * 16;  // 0 or 16
    const int brow  = tid >> 2;        // 0..31
    const int bcol0 = (tid & 3) * 16;  // 0,16,32,48

    for (int kc = 0; kc < K; kc += 32) {
        {   // stage A tile (64x32) fp32 -> f16
            const int grow = rb + arow;
            const float* src = (grow < Msplit) ? (A0 + (size_t)grow * K)
                                               : (A1 + (size_t)(grow - Msplit) * K);
            #pragma unroll
            for (int c = 0; c < 16; ++c)
                As[arow][acol0 + c] = (_Float16)src[kc + acol0 + c];
        }
        {   // stage B tile (32x64)
            #pragma unroll
            for (int c = 0; c < 16; ++c)
                Bs[brow][bcol0 + c] = (_Float16)B[(size_t)(kc + brow) * N + cb + bcol0 + c];
        }
        __syncthreads();

        v16h af;
        #pragma unroll
        for (int e = 0; e < 16; ++e)
            af[e] = As[wave * 16 + mloc][a_kk(e, half)];

        #pragma unroll
        for (int ct = 0; ct < 4; ++ct) {
            v16h bf;
            #pragma unroll
            for (int e = 0; e < 16; ++e)
                bf[e] = Bs[b_kk(e, half)][ct * 16 + mloc];
            acc[ct] = __builtin_amdgcn_wmma_f32_16x16x32_f16(
                false, af, false, bf, (short)0, acc[ct], false, false);
        }
        __syncthreads();
    }

    // C layout: vgpr v -> row v + 8*half, lane&15 -> col
    #pragma unroll
    for (int ct = 0; ct < 4; ++ct)
        #pragma unroll
        for (int v = 0; v < 8; ++v) {
            const int row = rb + wave * 16 + v + 8 * half;
            const int col = cb + ct * 16 + mloc;
            if (OUT_HALF)
                ((_Float16*)Cout)[(size_t)row * N + col] = (_Float16)acc[ct][v];
            else
                ((float*)Cout)[(size_t)row * N + col] = acc[ct][v];
        }
}

// ---------------------------------------------------------------------------
// Flash attention with Transformer-XL relative shift.
// One wave (32 thr) per block; block = (i_tile of 16 rows, head h, batch b).
// BD_shifted[i,j] = (q_i + rrb) . rk[j - i + 1023]  (masked region only)
// V tile staged LDS-side via global_load_async_to_lds_b128 (ASYNCcnt),
// issued before the AC/BD WMMAs so the DMA overlaps matrix work.
// ---------------------------------------------------------------------------
__global__ __launch_bounds__(32) void attn_kernel(
    const _Float16* __restrict__ qkv,  // (SEG*BSZ, 3072) f16, row = s*BSZ+b
    const _Float16* __restrict__ rk,   // (SEG, 1024) f16
    const float* __restrict__ rwb,     // (16,64)
    const float* __restrict__ rrb,     // (16,64)
    float* __restrict__ av)            // (TOT*BSZ, 1024) f32
{
    const int i0   = blockIdx.x * 16;
    const int h    = blockIdx.y;
    const int b    = blockIdx.z;
    const int lane = threadIdx.x;
    const int m    = lane & 15;
    const int half = lane >> 4;
    const float scale = 0.125f;        // 1/sqrt(64)

    __shared__ float    bd_s[16][48];
    __shared__ _Float16 p_s[16][32];
    __shared__ _Float16 vs[32][68];    // V tile, 136B row stride (padded)

    // ---- build q fragments (q+rw_bias and q+rr_bias), K = d in 2 chunks of 32 ----
    v16h aw[2], ar[2];
    const size_t qrow = (size_t)((MEM_LEN + i0 + m) * BSZ + b) * 3072 + h * 64;
    #pragma unroll
    for (int c = 0; c < 2; ++c)
        #pragma unroll
        for (int e = 0; e < 16; ++e) {
            const int d = c * 32 + a_kk(e, half);
            const float qv = (float)qkv[qrow + d];
            aw[c][e] = (_Float16)(qv + rwb[h * 64 + d]);
            ar[c][e] = (_Float16)(qv + rrb[h * 64 + d]);
        }

    float rmax[8], rsum[8];
    v8f acc[4];
    #pragma unroll
    for (int v = 0; v < 8; ++v) { rmax[v] = -1e30f; rsum[v] = 0.0f; }
    #pragma unroll
    for (int dt = 0; dt < 4; ++dt)
        #pragma unroll
        for (int v = 0; v < 8; ++v) acc[dt][v] = 0.0f;

    const int jend = i0 + 15 + MEM_LEN;   // last j any row of this tile may see
    for (int j0 = 0; j0 <= jend; j0 += 32) {
        // ------- async-stage V tile (32 x 64 f16) into LDS; overlaps AC/BD -------
        // WAR guard: previous iteration's ds reads of vs must drain first.
        asm volatile("s_wait_dscnt 0" ::: "memory");
        {
            const int chunk = lane & 7;       // 16B chunk within a 128B row
            const int rsub  = lane >> 3;      // 0..3
            #pragma unroll
            for (int t = 0; t < 8; ++t) {
                const int row = t * 4 + rsub; // 0..31
                const int j   = j0 + row;
                const int jc  = (j < SEG_LEN) ? j : (SEG_LEN - 1);
                const _Float16* gp = qkv + (size_t)(jc * BSZ + b) * 3072 + 2048
                                   + h * 64 + chunk * 8;
                const unsigned loff = (unsigned)(uintptr_t)&vs[row][chunk * 8];
                asm volatile("global_load_async_to_lds_b128 %0, %1, off"
                             :: "v"(loff),
                                "v"((unsigned long long)(uintptr_t)gp)
                             : "memory");
            }
        }

        // ---------------- AC = (q+rwb) . k^T  (16 x 32) ----------------
        v8f s0, s1;
        #pragma unroll
        for (int v = 0; v < 8; ++v) { s0[v] = 0.0f; s1[v] = 0.0f; }
        #pragma unroll
        for (int ct = 0; ct < 2; ++ct) {
            const int j  = j0 + ct * 16 + m;
            const int jc = (j < SEG_LEN) ? j : (SEG_LEN - 1);
            #pragma unroll
            for (int c = 0; c < 2; ++c) {
                // B-frag: 16 contiguous halves (32B aligned)
                const _Float16* kb = qkv + (size_t)(jc * BSZ + b) * 3072 + 1024
                                   + h * 64 + c * 32 + 16 * half;
                v16h bf = *(const v16h*)kb;
                if (ct == 0)
                    s0 = __builtin_amdgcn_wmma_f32_16x16x32_f16(false, aw[c], false, bf,
                                                                (short)0, s0, false, false);
                else
                    s1 = __builtin_amdgcn_wmma_f32_16x16x32_f16(false, aw[c], false, bf,
                                                                (short)0, s1, false, false);
            }
        }

        // ------------- BD band: (q+rrb) . rk[t0 .. t0+47]^T (16 x 48) -------------
        const int t0 = j0 + 1008 - i0;    // j0 + 1023 - i0 - 15
        v8f sb[3];
        #pragma unroll
        for (int ct = 0; ct < 3; ++ct)
            #pragma unroll
            for (int v = 0; v < 8; ++v) sb[ct][v] = 0.0f;
        #pragma unroll
        for (int ct = 0; ct < 3; ++ct) {
            int t = t0 + ct * 16 + m;
            t = (t < SEG_LEN) ? t : (SEG_LEN - 1);   // t >= 0 always
            #pragma unroll
            for (int c = 0; c < 2; ++c) {
                const _Float16* rbp = rk + (size_t)t * 1024 + h * 64 + c * 32 + 16 * half;
                v16h bf = *(const v16h*)rbp;
                sb[ct] = __builtin_amdgcn_wmma_f32_16x16x32_f16(false, ar[c], false, bf,
                                                                (short)0, sb[ct], false, false);
            }
        }
        // spill band to LDS for the per-row diagonal gather
        #pragma unroll
        for (int ct = 0; ct < 3; ++ct)
            #pragma unroll
            for (int v = 0; v < 8; ++v)
                bd_s[v + 8 * half][ct * 16 + m] = sb[ct][v];
        asm volatile("s_wait_dscnt 0" ::: "memory");

        // -------- combine, mask, online softmax (rows m = v + 8*half) --------
        float sc0[8], sc1[8], nm[8];
        #pragma unroll
        for (int v = 0; v < 8; ++v) {
            const int mm = v + 8 * half;
            const int ii = i0 + mm;
            const int ja = j0 + m, jb2 = j0 + 16 + m;
            float x0 = (s0[v] + bd_s[mm][m - mm + 15]) * scale;
            float x1 = (s1[v] + bd_s[mm][m + 16 - mm + 15]) * scale;
            if (ja  > ii + MEM_LEN || ja  >= SEG_LEN) x0 = -1e30f;
            if (jb2 > ii + MEM_LEN || jb2 >= SEG_LEN) x1 = -1e30f;
            sc0[v] = x0; sc1[v] = x1;
            float mx = fmaxf(x0, x1);
            #pragma unroll
            for (int off = 1; off < 16; off <<= 1)
                mx = fmaxf(mx, __shfl_xor(mx, off, 32));
            nm[v] = fmaxf(rmax[v], mx);
        }
        #pragma unroll
        for (int v = 0; v < 8; ++v) {
            const float f  = __expf(rmax[v] - nm[v]);
            const float p0 = __expf(sc0[v] - nm[v]);
            const float p1 = __expf(sc1[v] - nm[v]);
            float ls = p0 + p1;
            #pragma unroll
            for (int off = 1; off < 16; off <<= 1)
                ls += __shfl_xor(ls, off, 32);
            rsum[v] = rsum[v] * f + ls;
            rmax[v] = nm[v];
            #pragma unroll
            for (int dt = 0; dt < 4; ++dt) acc[dt][v] *= f;
            const int mm = v + 8 * half;
            p_s[mm][m]      = (_Float16)p0;
            p_s[mm][m + 16] = (_Float16)p1;
        }
        asm volatile("s_wait_dscnt 0" ::: "memory");

        // ---------------- PV: acc += P (16x32) * V (32x64) ----------------
        v16h pf;
        #pragma unroll
        for (int e = 0; e < 16; ++e)
            pf[e] = p_s[m][a_kk(e, half)];

        // V tile must be resident in LDS now
        asm volatile("s_wait_asynccnt 0" ::: "memory");
        #pragma unroll
        for (int dt = 0; dt < 4; ++dt) {
            v16h bf;
            #pragma unroll
            for (int e = 0; e < 16; ++e)
                bf[e] = vs[b_kk(e, half)][dt * 16 + m];
            acc[dt] = __builtin_amdgcn_wmma_f32_16x16x32_f16(false, pf, false, bf,
                                                             (short)0, acc[dt], false, false);
        }
    }

    // ---- epilogue: normalize and write attn_vec (row = i*BSZ + b, col = h*64+d) ----
    #pragma unroll
    for (int dt = 0; dt < 4; ++dt)
        #pragma unroll
        for (int v = 0; v < 8; ++v) {
            const int mm  = v + 8 * half;
            const int row = (i0 + mm) * BSZ + b;
            av[(size_t)row * 1024 + h * 64 + dt * 16 + m] = acc[dt][v] / rsum[v];
        }
}

// ---------------------------------------------------------------------------
extern "C" void kernel_launch(void* const* d_in, const int* in_sizes, int n_in,
                              void* d_out, int out_size, void* d_ws, size_t ws_size,
                              hipStream_t stream) {
    (void)in_sizes; (void)n_in; (void)out_size; (void)ws_size;
    const float* w    = (const float*)d_in[0];
    const float* r    = (const float*)d_in[1];
    const float* rwb  = (const float*)d_in[2];
    const float* rrb  = (const float*)d_in[3];
    // d_in[4] = attn_mask (bool) — mask computed analytically, unused
    const float* mems = (const float*)d_in[5];
    const float* Wqkv = (const float*)d_in[6];
    const float* Wr   = (const float*)d_in[7];
    const float* Wo   = (const float*)d_in[8];
    float* out = (float*)d_out;

    // workspace: qkv f16 (8192x3072) | rk f16 (2048x1024) | attn_vec f32 (4096x1024)
    _Float16* qkv_h = (_Float16*)d_ws;
    _Float16* rk_h  = qkv_h + (size_t)8192 * 3072;
    float*    av    = (float*)(rk_h + (size_t)2048 * 1024);

    // 1) QKV projection: [mems; w] (8192x1024) @ Wqkv (1024x3072) -> f16
    gemm_f16_kernel<true><<<dim3(3072 / 64, 8192 / 64), 128, 0, stream>>>(
        mems, w, MEM_LEN * BSZ, Wqkv, qkv_h, 8192, 3072, 1024);

    // 2) r projection: r (2048x1024) @ Wr (1024x1024) -> f16
    gemm_f16_kernel<true><<<dim3(1024 / 64, 2048 / 64), 128, 0, stream>>>(
        r, r, SEG_LEN, Wr, rk_h, 2048, 1024, 1024);

    // 3) flash attention with rel-shift; one wave per (i_tile, head, batch)
    attn_kernel<<<dim3(TOT_LEN / 16, N_HEAD, BSZ), 32, 0, stream>>>(
        qkv_h, rk_h, rwb, rrb, av);

    // 4) output projection: attn_vec (4096x1024) @ Wo (1024x1024) -> f32 out
    gemm_f16_kernel<false><<<dim3(1024 / 64, 4096 / 64), 128, 0, stream>>>(
        av, av, TOT_LEN * BSZ, Wo, out, 4096, 1024, 1024);
}
